// MambaPredictor_28028956573916
// MI455X (gfx1250) — matrix operand order repaired and checked by reading, not compiled, not run on hardware
//
#include <hip/hip_runtime.h>
#include <stdint.h>

// ---------------- problem constants ----------------
#define B_      8
#define L_      2048
#define DMODEL  512
#define DSTATE  16
#define DCONV   4
#define DINNER  1024
#define DTRANK  32
#define MTOK    (B_ * L_)          // 16384 tokens

typedef unsigned short u16;
typedef unsigned int   u32;
typedef __attribute__((ext_vector_type(16))) u16    v16u;
typedef __attribute__((ext_vector_type(16))) __bf16 v16bf;
typedef __attribute__((ext_vector_type(8)))  float  v8f;

// ---------------- helpers ----------------
__device__ __forceinline__ u16 f32_to_bf16_bits(float f) {
    u32 u = __builtin_bit_cast(u32, f);
    u32 r = u + 0x7FFFu + ((u >> 16) & 1u);   // round-to-nearest-even
    return (u16)(r >> 16);
}
__device__ __forceinline__ float bf16_bits_to_f32(u16 h) {
    u32 u = ((u32)h) << 16;
    return __builtin_bit_cast(float, u);
}
__device__ __forceinline__ float sigmoidf_(float x) {
    return 1.0f / (1.0f + __expf(-x));
}

// CDNA5 async memory->LDS DMA (tracked by ASYNCcnt, no VGPR data path).
// Generic pointer to __shared__ carries the LDS offset in its low 32 bits.
__device__ __forceinline__ void async_ld_b128(void* lds, const void* gaddr) {
    u32 loff = (u32)(uintptr_t)lds;
    asm volatile("global_load_async_to_lds_b128 %0, %1, off"
                 :: "v"(loff), "v"(gaddr) : "memory");
}
__device__ __forceinline__ void wait_async_le4() {   // prev tile done, next in flight
    asm volatile("s_wait_asynccnt 4" ::: "memory");
}
__device__ __forceinline__ void wait_async_le0() {
    asm volatile("s_wait_asynccnt 0" ::: "memory");
}

// ---------------- fp32 -> bf16 conversion (contiguous) ----------------
__global__ __launch_bounds__(256) void cvt_f32_bf16(const float* __restrict__ src,
                                                    u16* __restrict__ dst, int n) {
    int i = blockIdx.x * 256 + threadIdx.x;
    if (i < n) dst[i] = f32_to_bf16_bits(src[i]);
}

// fp32[16384][64] cols 0..31 -> bf16[16384][32]   (dt-rank slice of x_dbl)
__global__ __launch_bounds__(256) void cvt_dtr(const float* __restrict__ xdbl,
                                               u16* __restrict__ dst) {
    int i = blockIdx.x * 256 + threadIdx.x;       // 0 .. 16384*32-1
    int row = i >> 5, col = i & 31;
    dst[i] = f32_to_bf16_bits(xdbl[(size_t)row * 64 + col]);
}

// ---------------- tiled bf16 WMMA GEMM, double-buffered async-LDS ----------------
// C[M,N] = A[M,K] * B^T (A bf16 [M,K], B bf16 row-major [N,K])
// Block: 64x64 tile, 128 threads = 4 waves, wave -> 32x32 subtile (2x2 frags).
// EPI: 0 = store f32 | 1 = store bf16 | 2 = softplus(acc + bias[n]) -> f32
template <int EPI>
__global__ __launch_bounds__(128) void gemm_wmma(
    const u16*  __restrict__ Ap, int lda,
    const u16*  __restrict__ Bp, int ldb,
    void*       __restrict__ Cp, int ldc,
    const float* __restrict__ bias, int K)
{
    __shared__ u16 As[2][64][40];   // 80B row stride (16B-aligned), spreads banks
    __shared__ u16 Bs[2][64][40];

    const int tid   = threadIdx.x;
    const int wave  = tid >> 5;
    const int lane  = tid & 31;
    const int m0    = blockIdx.y << 6;
    const int n0    = blockIdx.x << 6;
    const int wm    = (wave >> 1) << 5;   // 0 / 32
    const int wn    = (wave & 1) << 5;    // 0 / 32
    const int row16 = lane & 15;
    const int half  = lane >> 4;

    // staging geometry: thread -> (row r & r+32, 16B segment c8) of both tiles
    const int r  = tid >> 2;
    const int c8 = (tid & 3) * 8;

    v8f acc[2][2];
    #pragma unroll
    for (int i = 0; i < 2; ++i)
        #pragma unroll
        for (int j = 0; j < 2; ++j)
            #pragma unroll
            for (int e = 0; e < 8; ++e) acc[i][j][e] = 0.0f;

    // 4 unconditional async DMAs per thread per tile (no EXEC predication):
    auto issue_tile = [&](int buf, int k0) {
        async_ld_b128(&As[buf][r][c8],      Ap + (size_t)(m0 + r)      * lda + k0 + c8);
        async_ld_b128(&As[buf][r + 32][c8], Ap + (size_t)(m0 + r + 32) * lda + k0 + c8);
        async_ld_b128(&Bs[buf][r][c8],      Bp + (size_t)(n0 + r)      * ldb + k0 + c8);
        async_ld_b128(&Bs[buf][r + 32][c8], Bp + (size_t)(n0 + r + 32) * ldb + k0 + c8);
    };

    const int ntiles = K >> 5;
    issue_tile(0, 0);                                 // prologue: tile 0

    for (int kt = 0; kt < ntiles; ++kt) {
        const int cur = kt & 1;
        if (kt + 1 < ntiles) {
            issue_tile(cur ^ 1, (kt + 1) << 5);       // prefetch next tile
            wait_async_le4();                         // current tile's 4 DMAs done
        } else {
            wait_async_le0();
        }
        __syncthreads();                              // all waves' segments landed

        // A frag (16x32, ISA layout): lane m = lane%16, half = lane/16
        // elems 0..7 -> K = half*8 + 0..7 ; elems 8..15 -> K = 16 + half*8 + 0..7
        v16bf afrag[2], bfrag[2];
        #pragma unroll
        for (int tm = 0; tm < 2; ++tm) {
            const u16* p0 = &As[cur][wm + tm * 16 + row16][half * 8];
            const u16* p1 = &As[cur][wm + tm * 16 + row16][16 + half * 8];
            v16u t;
            #pragma unroll
            for (int e = 0; e < 8; ++e) { t[e] = p0[e]; t[e + 8] = p1[e]; }
            afrag[tm] = __builtin_bit_cast(v16bf, t);
        }
        // B frag (32x16): lane n = lane%16, elems 0..7 -> K = half*8 + 0..7 ; elems 8..15 -> K = 16 + half*8 + 0..7
        #pragma unroll
        for (int tn = 0; tn < 2; ++tn) {
            const u16* p0 = &Bs[cur][wn + tn * 16 + row16][half * 8];
            const u16* p1 = &Bs[cur][wn + tn * 16 + row16][16 + half * 8];
            v16u t;
            #pragma unroll
            for (int e = 0; e < 8; ++e) { t[e] = p0[e]; t[e + 8] = p1[e]; }
            bfrag[tn] = __builtin_bit_cast(v16bf, t);
        }

        #pragma unroll
        for (int tm = 0; tm < 2; ++tm)
            #pragma unroll
            for (int tn = 0; tn < 2; ++tn)
                acc[tm][tn] = __builtin_amdgcn_wmma_f32_16x16x32_bf16(
                    false, afrag[tm], false, bfrag[tn], (short)0, acc[tm][tn],
                    false, false);
        __syncthreads();                              // reads done before rewrite
    }

    // epilogue; D layout: VGPR r -> m = r + 8*half, col n = lane%16
    #pragma unroll
    for (int tm = 0; tm < 2; ++tm)
        #pragma unroll
        for (int tn = 0; tn < 2; ++tn) {
            int gn = n0 + wn + tn * 16 + row16;
            #pragma unroll
            for (int rr = 0; rr < 8; ++rr) {
                int gm = m0 + wm + tm * 16 + rr + half * 8;
                float v = acc[tm][tn][rr];
                if (EPI == 0) {
                    ((float*)Cp)[(size_t)gm * ldc + gn] = v;
                } else if (EPI == 1) {
                    ((u16*)Cp)[(size_t)gm * ldc + gn] = f32_to_bf16_bits(v);
                } else {
                    v += bias[gn];
                    v = (v > 20.0f) ? v : log1pf(__expf(v));  // softplus
                    ((float*)Cp)[(size_t)gm * ldc + gn] = v;
                }
            }
        }
}

// ---------------- depthwise causal conv (k=4) + SiLU ----------------
__global__ __launch_bounds__(256) void conv_silu(const u16* __restrict__ xraw,
                                                 const float* __restrict__ cw,
                                                 const float* __restrict__ cb,
                                                 u16* __restrict__ xconv)
{
    int idx = blockIdx.x * 256 + threadIdx.x;     // flat (b,l,d), exact count
    int d = idx & (DINNER - 1);
    int l = (idx >> 10) & (L_ - 1);
    float acc = cb[d];
    #pragma unroll
    for (int j = 0; j < DCONV; ++j) {
        int ls = l + j - (DCONV - 1);
        if (ls >= 0)
            acc += cw[d * DCONV + j] * bf16_bits_to_f32(xraw[idx + (j - (DCONV - 1)) * DINNER]);
    }
    float s = acc * sigmoidf_(acc);               // SiLU
    xconv[idx] = f32_to_bf16_bits(s);
}

// ---------------- selective scan (serial over L) ----------------
// One thread per (b, d) channel; 16 states in registers; C only at t = L-1.
__global__ __launch_bounds__(256) void mamba_scan(const float* __restrict__ dt,
                                                  const u16*  __restrict__ xconv,
                                                  const float* __restrict__ xdbl,
                                                  const float* __restrict__ A_log,
                                                  const float* __restrict__ Dv,
                                                  float* __restrict__ y_pre)
{
    int tid = blockIdx.x * 256 + threadIdx.x;     // 0..8191
    int b = tid >> 10, d = tid & (DINNER - 1);

    float A[DSTATE], h[DSTATE];
    #pragma unroll
    for (int s = 0; s < DSTATE; ++s) { A[s] = -__expf(A_log[d * DSTATE + s]); h[s] = 0.0f; }

    size_t rowbase = (size_t)b * L_;
    for (int t = 0; t < L_; ++t) {
        size_t row = rowbase + t;
        if (t + 8 < L_) {                          // hide dependent-load latency
            __builtin_prefetch(&dt[(row + 8) * DINNER + d], 0, 1);
            __builtin_prefetch(&xconv[(row + 8) * DINNER + d], 0, 1);
        }
        float dtv = dt[row * DINNER + d];
        float xv  = bf16_bits_to_f32(xconv[row * DINNER + d]);
        const float* Bm = &xdbl[row * 64 + DTRANK];           // cols 32..47
        float dx = dtv * xv;
        #pragma unroll
        for (int s = 0; s < DSTATE; ++s)
            h[s] = __expf(dtv * A[s]) * h[s] + dx * Bm[s];
    }
    size_t lrow = rowbase + (L_ - 1);
    const float* Cm = &xdbl[lrow * 64 + DTRANK + DSTATE];     // cols 48..63
    float y = 0.0f;
    #pragma unroll
    for (int s = 0; s < DSTATE; ++s) y += h[s] * Cm[s];
    y += bf16_bits_to_f32(xconv[lrow * DINNER + d]) * Dv[d];  // skip connection
    y_pre[tid] = y;
}

// ---------------- z-gate at last token ----------------
__global__ __launch_bounds__(256) void zgate(const float* __restrict__ P,
                                             const float* __restrict__ W_in,
                                             const float* __restrict__ y_pre,
                                             float* __restrict__ y_g)
{
    int tid = blockIdx.x * 256 + threadIdx.x;     // 0..8191
    int b = tid >> 10, d = tid & (DINNER - 1);
    const float* p = &P[((size_t)b * L_ + (L_ - 1)) * DMODEL];
    const float* w = &W_in[(size_t)(DINNER + d) * DMODEL];    // z-half of W_in
    float z = 0.0f;
    for (int k = 0; k < DMODEL; ++k) z += p[k] * w[k];
    y_g[tid] = y_pre[tid] * (z * sigmoidf_(z));
}

// ---------------- out-projection at last token ----------------
__global__ __launch_bounds__(256) void outproj(const float* __restrict__ y_g,
                                               const float* __restrict__ W_out,
                                               float* __restrict__ out_last)
{
    int tid = blockIdx.x * 256 + threadIdx.x;     // 0..4095
    int b = tid >> 9, e = tid & (DMODEL - 1);
    const float* y = &y_g[(size_t)b * DINNER];
    const float* w = &W_out[(size_t)e * DINNER];
    float acc = 0.0f;
    for (int k = 0; k < DINNER; ++k) acc += y[k] * w[k];
    out_last[tid] = acc;
}

// ---------------- LayerNorm + head ----------------
__global__ __launch_bounds__(256) void ln_head(const float* __restrict__ out_last,
                                               const float* __restrict__ g,
                                               const float* __restrict__ beta,
                                               const float* __restrict__ Wh,
                                               const float* __restrict__ bh,
                                               float* __restrict__ out)
{
    __shared__ float red[256];
    __shared__ float xn[DMODEL];
    int b = blockIdx.x, t = threadIdx.x;
    const float* x = out_last + (size_t)b * DMODEL;
    float v0 = x[t], v1 = x[t + 256];
    red[t] = v0 + v1;
    __syncthreads();
    for (int s = 128; s > 0; s >>= 1) { if (t < s) red[t] += red[t + s]; __syncthreads(); }
    float mu = red[0] * (1.0f / DMODEL);
    __syncthreads();
    float d0 = v0 - mu, d1 = v1 - mu;
    red[t] = d0 * d0 + d1 * d1;
    __syncthreads();
    for (int s = 128; s > 0; s >>= 1) { if (t < s) red[t] += red[t + s]; __syncthreads(); }
    float rstd = rsqrtf(red[0] * (1.0f / DMODEL) + 1e-5f);
    xn[t]       = d0 * rstd * g[t]       + beta[t];
    xn[t + 256] = d1 * rstd * g[t + 256] + beta[t + 256];
    __syncthreads();
    for (int e = t; e < DMODEL; e += 256) {
        const float* w = Wh + (size_t)e * DMODEL;
        float acc = bh[e];
        for (int k = 0; k < DMODEL; ++k) acc += xn[k] * w[k];
        out[(size_t)b * DMODEL + e] = acc;
    }
}

// ---------------- workspace layout (bytes, 256-aligned) ----------------
#define OFF_WINX   ((size_t)0)           // 1024*512  bf16 =  1,048,576
#define OFF_WX     ((size_t)1048576)     // 64*1024   bf16 =    131,072
#define OFF_WDT    ((size_t)1179648)     // 1024*32   bf16 =     65,536
#define OFF_PHB    ((size_t)1245184)     // 16384*512 bf16 = 16,777,216
#define OFF_XRAW   ((size_t)18022400)    // 16384*1024 bf16 = 33,554,432
#define OFF_XCONV  ((size_t)51576832)    // 33,554,432
#define OFF_XDBL   ((size_t)85131264)    // 16384*64 f32  =  4,194,304
#define OFF_DTR    ((size_t)89325568)    // 16384*32 bf16 =  1,048,576
#define OFF_DT     ((size_t)90374144)    // 16384*1024 f32 = 67,108,864
#define OFF_YPRE   ((size_t)157483008)   // 8192 f32
#define OFF_YGATE  ((size_t)157515776)   // 8192 f32
#define OFF_OUTL   ((size_t)157548544)   // 4096 f32

extern "C" void kernel_launch(void* const* d_in, const int* in_sizes, int n_in,
                              void* d_out, int out_size, void* d_ws, size_t ws_size,
                              hipStream_t stream)
{
    (void)in_sizes; (void)n_in; (void)out_size; (void)ws_size;
    const float* p_hist = (const float*)d_in[0];
    const float* W_in   = (const float*)d_in[1];
    const float* conv_w = (const float*)d_in[2];
    const float* conv_b = (const float*)d_in[3];
    const float* W_x    = (const float*)d_in[4];
    const float* W_dt   = (const float*)d_in[5];
    const float* b_dt   = (const float*)d_in[6];
    const float* A_log  = (const float*)d_in[7];
    const float* Dvec   = (const float*)d_in[8];
    const float* W_out  = (const float*)d_in[9];
    const float* ln_g   = (const float*)d_in[10];
    const float* ln_b   = (const float*)d_in[11];
    const float* W_head = (const float*)d_in[12];
    const float* b_head = (const float*)d_in[13];
    float* out = (float*)d_out;

    char* ws = (char*)d_ws;
    u16*   winx   = (u16*)(ws + OFF_WINX);
    u16*   wx     = (u16*)(ws + OFF_WX);
    u16*   wdt    = (u16*)(ws + OFF_WDT);
    u16*   phb    = (u16*)(ws + OFF_PHB);
    u16*   xraw   = (u16*)(ws + OFF_XRAW);
    u16*   xconv  = (u16*)(ws + OFF_XCONV);
    float* xdbl   = (float*)(ws + OFF_XDBL);
    u16*   dtr    = (u16*)(ws + OFF_DTR);
    float* dtbuf  = (float*)(ws + OFF_DT);
    float* ypre   = (float*)(ws + OFF_YPRE);
    float* ygate  = (float*)(ws + OFF_YGATE);
    float* outl   = (float*)(ws + OFF_OUTL);

    // 1) bf16 conversions: weights + activations (p_history)
    cvt_f32_bf16<<<(DINNER * DMODEL + 255) / 256, 256, 0, stream>>>(W_in, winx, DINNER * DMODEL);
    cvt_f32_bf16<<<((DTRANK + 2 * DSTATE) * DINNER + 255) / 256, 256, 0, stream>>>(W_x, wx, (DTRANK + 2 * DSTATE) * DINNER);
    cvt_f32_bf16<<<(DINNER * DTRANK + 255) / 256, 256, 0, stream>>>(W_dt, wdt, DINNER * DTRANK);
    cvt_f32_bf16<<<(MTOK * DMODEL + 255) / 256, 256, 0, stream>>>(p_hist, phb, MTOK * DMODEL);

    // 2) in-projection x-branch: [16384,512] x [512,1024]^T -> bf16
    gemm_wmma<1><<<dim3(DINNER / 64, MTOK / 64), 128, 0, stream>>>(
        phb, DMODEL, winx, DMODEL, xraw, DINNER, nullptr, DMODEL);

    // 3) depthwise causal conv + SiLU
    conv_silu<<<(MTOK * DINNER) / 256, 256, 0, stream>>>(xraw, conv_w, conv_b, xconv);

    // 4) x_dbl = xconv @ W_x^T : [16384,1024] x [1024,64]^T -> f32
    gemm_wmma<0><<<dim3(1, MTOK / 64), 128, 0, stream>>>(
        xconv, DINNER, wx, DINNER, xdbl, 64, nullptr, DINNER);

    // 5a) bf16 slice of dt-rank part of x_dbl
    cvt_dtr<<<(MTOK * DTRANK) / 256, 256, 0, stream>>>(xdbl, dtr);

    // 5b) dt = softplus(dtr @ W_dt^T + b_dt) : K=32 -> f32
    gemm_wmma<2><<<dim3(DINNER / 64, MTOK / 64), 128, 0, stream>>>(
        dtr, DTRANK, wdt, DTRANK, dtbuf, DINNER, b_dt, DTRANK);

    // 6) selective scan (serial over L), outputs y at final token only
    mamba_scan<<<(B_ * DINNER) / 256, 256, 0, stream>>>(dtbuf, xconv, xdbl, A_log, Dvec, ypre);

    // 7) gate with silu(z) at last token
    zgate<<<(B_ * DINNER) / 256, 256, 0, stream>>>(p_hist, W_in, ypre, ygate);

    // 8) out-projection at last token
    outproj<<<(B_ * DMODEL) / 256, 256, 0, stream>>>(ygate, W_out, outl);

    // 9) LayerNorm + head
    ln_head<<<B_, 256, 0, stream>>>(outl, ln_g, ln_b, W_head, b_head, out);
}
